// CfdGnnPool_49744311222702
// MI455X (gfx1250) — matrix operand order repaired and checked by reading, not compiled, not run on hardware
//
#include <hip/hip_runtime.h>
#include <hip/hip_bf16.h>
#include <math.h>

// ---------------- problem constants (from reference) ----------------
#define N_GRAPHS        4
#define NODES_PER_GRAPH 8192
#define N_NODES         (N_GRAPHS * NODES_PER_GRAPH)   // 32768
#define N_EDGES         (N_NODES * 16)                 // 524288
#define HID             128
#define K_OUT           512
#define WAVES_PER_BLOCK 4

typedef __attribute__((ext_vector_type(16))) __bf16 v16bf;
typedef __attribute__((ext_vector_type(8)))  __bf16 bf16x8;
typedef __attribute__((ext_vector_type(8)))  float  v8f;

// ---------------------------------------------------------------
// generic fill
// ---------------------------------------------------------------
__global__ void fill_kernel(float* __restrict__ p, size_t n, float v) {
    size_t i = (size_t)blockIdx.x * blockDim.x + threadIdx.x;
    if (i < n) p[i] = v;
}

// ---------------------------------------------------------------
// h = x @ proj_w + proj_b + sincos(mesh_pos); also bf16 copy of h
// ---------------------------------------------------------------
__global__ void node_init_kernel(const float* __restrict__ x,
                                 const float* __restrict__ pos,
                                 const float* __restrict__ pw,
                                 const float* __restrict__ pb,
                                 float* __restrict__ h,
                                 __bf16* __restrict__ hbf) {
    size_t idx = (size_t)blockIdx.x * blockDim.x + threadIdx.x;
    if (idx >= (size_t)N_NODES * HID) return;
    int n = (int)(idx >> 7);
    int c = (int)(idx & 127);
    float acc = pb[c];
    #pragma unroll
    for (int i = 0; i < 3; ++i) acc += x[n * 3 + i] * pw[i * HID + c];
    float p  = (c < 64) ? pos[n * 2 + 0] : pos[n * 2 + 1];
    int   cc = c & 63;
    int   k  = cc & 31;
    float omega = expf(-((float)(2 * k) / 64.0f) * logf(10000.0f));
    float ang = p * omega;
    acc += (cc < 32) ? sinf(ang) : cosf(ang);
    h[idx]   = acc;
    hbf[idx] = (__bf16)acc;
}

// ---------------------------------------------------------------
// cnt[dst] += 1
// ---------------------------------------------------------------
__global__ void degree_kernel(const int* __restrict__ edges,
                              float* __restrict__ cnt) {
    int e = blockIdx.x * blockDim.x + threadIdx.x;
    if (e >= N_EDGES) return;
    int d = edges[2 * e + 1];
    unsafeAtomicAdd(&cnt[d], 1.0f);
}

// ---------------------------------------------------------------
// shared WMMA helpers
// ---------------------------------------------------------------
__device__ __forceinline__ v16bf make_a_frag(const __bf16* rp, int half) {
    // A (16x32 bf16): elems 0..15 -> K = half*16 + 0..15
    bf16x8 lo = *(const bf16x8*)(rp + half * 16);
    bf16x8 hi = *(const bf16x8*)(rp + half * 16 + 8);
    v16bf a;
    #pragma unroll
    for (int i = 0; i < 8; ++i) { a[i] = lo[i]; a[8 + i] = hi[i]; }
    return a;
}

__device__ __forceinline__ float silu(float v) {
    return v / (1.0f + __expf(-v));
}

#define WMMA_BF16(A, B, C) \
    __builtin_amdgcn_wmma_f32_16x16x32_bf16(false, (A), false, (B), (short)0, (C), false, false)

// ---------------------------------------------------------------
// Edge message MLP:  m = silu([h_dst,h_src] @ W1 + b1) @ W2 + b2
// scatter:           agg[dst] += m
// one wave per 32-edge pair-tile (2 x 16 rows), bf16 WMMA, f32 acc.
// B fragments double-buffered; each B feeds 2 WMMAs.
// ---------------------------------------------------------------
__global__ __launch_bounds__(WAVES_PER_BLOCK * 32)
void edge_mlp_kernel(const __bf16* __restrict__ hbf,
                     const int*    __restrict__ edges,
                     const float*  __restrict__ w1, const float* __restrict__ b1,
                     const float*  __restrict__ w2, const float* __restrict__ b2,
                     float* __restrict__ agg) {
    __shared__ __align__(32) __bf16 W1t[128][256];                   // 64 KB  [n][k]
    __shared__ __align__(32) __bf16 W2t[128][128];                   // 32 KB  [n][k]
    __shared__ __align__(32) __bf16 XP[WAVES_PER_BLOCK][32][128];    // 32 KB  layer1->layer2

    const int tid = threadIdx.x;
    for (int idx = tid; idx < 256 * 128; idx += blockDim.x) {
        int k = idx >> 7, n = idx & 127;
        W1t[n][k] = (__bf16)w1[idx];
    }
    for (int idx = tid; idx < 128 * 128; idx += blockDim.x) {
        int k = idx >> 7, n = idx & 127;
        W2t[n][k] = (__bf16)w2[idx];
    }
    __syncthreads();

    const int lane = tid & 31;
    const int wave = tid >> 5;
    const int m    = lane & 15;
    const int half = lane >> 4;

    float bias1[8], bias2[8];
    #pragma unroll
    for (int t = 0; t < 8; ++t) { bias1[t] = b1[t * 16 + m]; bias2[t] = b2[t * 16 + m]; }

    const int nPairs = N_EDGES / 32;                  // 16384
    const int stride = gridDim.x * WAVES_PER_BLOCK;
    for (int pair = blockIdx.x * WAVES_PER_BLOCK + wave; pair < nPairs; pair += stride) {
        const int e0 = pair * 32;
        const int s0 = edges[2 * (e0 + m) + 0];
        const int d0 = edges[2 * (e0 + m) + 1];
        const int s1 = edges[2 * (e0 + 16 + m) + 0];
        const int d1 = edges[2 * (e0 + 16 + m) + 1];
        const __bf16* rowd0 = hbf + (size_t)d0 * HID;
        const __bf16* rows0 = hbf + (size_t)s0 * HID;
        const __bf16* rowd1 = hbf + (size_t)d1 * HID;
        const __bf16* rows1 = hbf + (size_t)s1 * HID;
        int dpre0[8], dpre1[8];
        #pragma unroll
        for (int r = 0; r < 8; ++r) {
            dpre0[r] = edges[2 * (e0 + r + half * 8) + 1];
            dpre1[r] = edges[2 * (e0 + 16 + r + half * 8) + 1];
        }

        // ---- layer 1: 2x [16 x 256] @ [256 x 128] ----
        v8f acc0[8], acc1[8];
        #pragma unroll
        for (int t = 0; t < 8; ++t)
            #pragma unroll
            for (int r = 0; r < 8; ++r) { acc0[t][r] = bias1[t]; acc1[t][r] = bias1[t]; }

        v16bf bcur = *(const v16bf*)&W1t[m][half * 16];   // (kb=0, t=0)
        #pragma unroll
        for (int kb = 0; kb < 8; ++kb) {
            const int f0 = kb * 32;
            const __bf16* rp0 = (f0 < 128) ? (rowd0 + f0) : (rows0 + (f0 - 128));
            const __bf16* rp1 = (f0 < 128) ? (rowd1 + f0) : (rows1 + (f0 - 128));
            v16bf a0 = make_a_frag(rp0, half);
            v16bf a1 = make_a_frag(rp1, half);
            #pragma unroll
            for (int t = 0; t < 8; ++t) {
                const bool last = (kb == 7) && (t == 7);
                v16bf bnxt;
                if (!last) {
                    const int tn = (t < 7) ? (t + 1) : 0;
                    const int fn = (t < 7) ? f0 : (f0 + 32);
                    bnxt = *(const v16bf*)&W1t[tn * 16 + m][fn + half * 16];
                }
                acc0[t] = WMMA_BF16(a0, bcur, acc0[t]);
                acc1[t] = WMMA_BF16(a1, bcur, acc1[t]);
                if (!last) bcur = bnxt;
            }
        }

        // ---- silu, transpose via per-wave LDS scratch ----
        #pragma unroll
        for (int t = 0; t < 8; ++t)
            #pragma unroll
            for (int r = 0; r < 8; ++r) {
                XP[wave][r + half * 8][t * 16 + m]      = (__bf16)silu(acc0[t][r]);
                XP[wave][16 + r + half * 8][t * 16 + m] = (__bf16)silu(acc1[t][r]);
            }
        asm volatile("s_wait_dscnt 0" ::: "memory");   // per-wave DS in-order; RAW guard

        // ---- layer 2: 2x [16 x 128] @ [128 x 128] ----
        v8f acc2a[8], acc2b[8];
        #pragma unroll
        for (int t = 0; t < 8; ++t)
            #pragma unroll
            for (int r = 0; r < 8; ++r) { acc2a[t][r] = bias2[t]; acc2b[t][r] = bias2[t]; }

        const __bf16* xr0 = &XP[wave][m][0];
        const __bf16* xr1 = &XP[wave][16 + m][0];
        bcur = *(const v16bf*)&W2t[m][half * 16];        // (kb=0, t=0)
        #pragma unroll
        for (int kb = 0; kb < 4; ++kb) {
            const int f0 = kb * 32;
            v16bf a0 = make_a_frag(xr0 + f0, half);
            v16bf a1 = make_a_frag(xr1 + f0, half);
            #pragma unroll
            for (int t = 0; t < 8; ++t) {
                const bool last = (kb == 3) && (t == 7);
                v16bf bnxt;
                if (!last) {
                    const int tn = (t < 7) ? (t + 1) : 0;
                    const int fn = (t < 7) ? f0 : (f0 + 32);
                    bnxt = *(const v16bf*)&W2t[tn * 16 + m][fn + half * 16];
                }
                acc2a[t] = WMMA_BF16(a0, bcur, acc2a[t]);
                acc2b[t] = WMMA_BF16(a1, bcur, acc2b[t]);
                if (!last) bcur = bnxt;
            }
        }

        // ---- scatter-add to agg[dst] ----
        #pragma unroll
        for (int t = 0; t < 8; ++t) {
            const int col = t * 16 + m;
            #pragma unroll
            for (int r = 0; r < 8; ++r) {
                unsafeAtomicAdd(&agg[(size_t)dpre0[r] * HID + col], acc2a[t][r]);
                unsafeAtomicAdd(&agg[(size_t)dpre1[r] * HID + col], acc2b[t][r]);
            }
        }
    }
}

// ---------------------------------------------------------------
// upd_in[n] = concat( bf16(h[n]), bf16(agg[n]/max(cnt,1)) )
// ---------------------------------------------------------------
__global__ void node_prep_kernel(const __bf16* __restrict__ hbf,
                                 const float*  __restrict__ agg,
                                 const float*  __restrict__ cnt,
                                 __bf16* __restrict__ upd_in) {
    size_t idx = (size_t)blockIdx.x * blockDim.x + threadIdx.x;
    if (idx >= (size_t)N_NODES * HID) return;
    int n = (int)(idx >> 7);
    int c = (int)(idx & 127);
    float inv = 1.0f / fmaxf(cnt[n], 1.0f);
    upd_in[(size_t)n * 256 + c]       = hbf[idx];
    upd_in[(size_t)n * 256 + 128 + c] = (__bf16)(agg[idx] * inv);
}

// ---------------------------------------------------------------
// Node update MLP: h_new = h + silu([h,agg] @ U1 + b1) @ U2 + b2
// one wave per 32-node pair-tile
// ---------------------------------------------------------------
__global__ __launch_bounds__(WAVES_PER_BLOCK * 32)
void update_mlp_kernel(const __bf16* __restrict__ upd_in,
                       const float*  __restrict__ h,
                       const float*  __restrict__ w1, const float* __restrict__ b1,
                       const float*  __restrict__ w2, const float* __restrict__ b2,
                       float* __restrict__ h_new) {
    __shared__ __align__(32) __bf16 W1t[128][256];
    __shared__ __align__(32) __bf16 W2t[128][128];
    __shared__ __align__(32) __bf16 XP[WAVES_PER_BLOCK][32][128];

    const int tid = threadIdx.x;
    for (int idx = tid; idx < 256 * 128; idx += blockDim.x) {
        int k = idx >> 7, n = idx & 127;
        W1t[n][k] = (__bf16)w1[idx];
    }
    for (int idx = tid; idx < 128 * 128; idx += blockDim.x) {
        int k = idx >> 7, n = idx & 127;
        W2t[n][k] = (__bf16)w2[idx];
    }
    __syncthreads();

    const int lane = tid & 31;
    const int wave = tid >> 5;
    const int m    = lane & 15;
    const int half = lane >> 4;

    float bias1[8], bias2[8];
    #pragma unroll
    for (int t = 0; t < 8; ++t) { bias1[t] = b1[t * 16 + m]; bias2[t] = b2[t * 16 + m]; }

    const int nPairs = N_NODES / 32;                  // 1024
    const int stride = gridDim.x * WAVES_PER_BLOCK;
    for (int pair = blockIdx.x * WAVES_PER_BLOCK + wave; pair < nPairs; pair += stride) {
        const int n0 = pair * 32;
        const __bf16* rowm0 = upd_in + (size_t)(n0 + m) * 256;
        const __bf16* rowm1 = upd_in + (size_t)(n0 + 16 + m) * 256;

        v8f acc0[8], acc1[8];
        #pragma unroll
        for (int t = 0; t < 8; ++t)
            #pragma unroll
            for (int r = 0; r < 8; ++r) { acc0[t][r] = bias1[t]; acc1[t][r] = bias1[t]; }

        v16bf bcur = *(const v16bf*)&W1t[m][half * 16];
        #pragma unroll
        for (int kb = 0; kb < 8; ++kb) {
            const int f0 = kb * 32;
            v16bf a0 = make_a_frag(rowm0 + f0, half);
            v16bf a1 = make_a_frag(rowm1 + f0, half);
            #pragma unroll
            for (int t = 0; t < 8; ++t) {
                const bool last = (kb == 7) && (t == 7);
                v16bf bnxt;
                if (!last) {
                    const int tn = (t < 7) ? (t + 1) : 0;
                    const int fn = (t < 7) ? f0 : (f0 + 32);
                    bnxt = *(const v16bf*)&W1t[tn * 16 + m][fn + half * 16];
                }
                acc0[t] = WMMA_BF16(a0, bcur, acc0[t]);
                acc1[t] = WMMA_BF16(a1, bcur, acc1[t]);
                if (!last) bcur = bnxt;
            }
        }

        #pragma unroll
        for (int t = 0; t < 8; ++t)
            #pragma unroll
            for (int r = 0; r < 8; ++r) {
                XP[wave][r + half * 8][t * 16 + m]      = (__bf16)silu(acc0[t][r]);
                XP[wave][16 + r + half * 8][t * 16 + m] = (__bf16)silu(acc1[t][r]);
            }
        asm volatile("s_wait_dscnt 0" ::: "memory");

        v8f acc2a[8], acc2b[8];
        #pragma unroll
        for (int t = 0; t < 8; ++t)
            #pragma unroll
            for (int r = 0; r < 8; ++r) { acc2a[t][r] = bias2[t]; acc2b[t][r] = bias2[t]; }

        const __bf16* xr0 = &XP[wave][m][0];
        const __bf16* xr1 = &XP[wave][16 + m][0];
        bcur = *(const v16bf*)&W2t[m][half * 16];
        #pragma unroll
        for (int kb = 0; kb < 4; ++kb) {
            const int f0 = kb * 32;
            v16bf a0 = make_a_frag(xr0 + f0, half);
            v16bf a1 = make_a_frag(xr1 + f0, half);
            #pragma unroll
            for (int t = 0; t < 8; ++t) {
                const bool last = (kb == 3) && (t == 7);
                v16bf bnxt;
                if (!last) {
                    const int tn = (t < 7) ? (t + 1) : 0;
                    const int fn = (t < 7) ? f0 : (f0 + 32);
                    bnxt = *(const v16bf*)&W2t[tn * 16 + m][fn + half * 16];
                }
                acc2a[t] = WMMA_BF16(a0, bcur, acc2a[t]);
                acc2b[t] = WMMA_BF16(a1, bcur, acc2b[t]);
                if (!last) bcur = bnxt;
            }
        }

        // residual write: h_new = h + mlp
        #pragma unroll
        for (int t = 0; t < 8; ++t) {
            const int col = t * 16 + m;
            #pragma unroll
            for (int r = 0; r < 8; ++r) {
                const size_t o0 = (size_t)(n0 + r + half * 8) * HID + col;
                const size_t o1 = (size_t)(n0 + 16 + r + half * 8) * HID + col;
                h_new[o0] = h[o0] + acc2a[t][r];
                h_new[o1] = h[o1] + acc2b[t][r];
            }
        }
    }
}

// ---------------------------------------------------------------
// nagg[dst] += h_new[src]   (4 columns per thread, float4 load)
// ---------------------------------------------------------------
__global__ void nagg_kernel(const int* __restrict__ edges,
                            const float* __restrict__ h_new,
                            float* __restrict__ nagg) {
    size_t idx = (size_t)blockIdx.x * blockDim.x + threadIdx.x;
    if (idx >= (size_t)N_EDGES * 32) return;
    int e  = (int)(idx >> 5);
    int c0 = (int)(idx & 31) * 4;
    int s = edges[2 * e + 0];
    int d = edges[2 * e + 1];
    float4 v = *(const float4*)(h_new + (size_t)s * HID + c0);
    float* dp = nagg + (size_t)d * HID + c0;
    unsafeAtomicAdd(dp + 0, v.x);
    unsafeAtomicAdd(dp + 1, v.y);
    unsafeAtomicAdd(dp + 2, v.z);
    unsafeAtomicAdd(dp + 3, v.w);
}

// ---------------------------------------------------------------
// score[n] = tanh( (nagg[n]/cnt) . rel_w + rel_b + h_new[n] . root_w )
// ---------------------------------------------------------------
__global__ void score_kernel(const float* __restrict__ nagg,
                             const float* __restrict__ cnt,
                             const float* __restrict__ h_new,
                             const float* __restrict__ rel_w,
                             const float* __restrict__ rel_b,
                             const float* __restrict__ root_w,
                             float* __restrict__ score) {
    int n = blockIdx.x * blockDim.x + threadIdx.x;
    if (n >= N_NODES) return;
    float inv = 1.0f / fmaxf(cnt[n], 1.0f);
    float s = rel_b[0];
    #pragma unroll 4
    for (int c = 0; c < HID; ++c) {
        s += nagg[(size_t)n * HID + c] * inv * rel_w[c];
        s += h_new[(size_t)n * HID + c] * root_w[c];
    }
    score[n] = tanhf(s);
}

// ---------------------------------------------------------------
// Exact top-k (stable, descending) via O(N^2) rank; one block/graph.
// rank(i) = #{j: s_j > s_i} + #{j < i: s_j == s_i}
// ---------------------------------------------------------------
__global__ __launch_bounds__(1024)
void topk_pool_kernel(const float* __restrict__ score,
                      const float* __restrict__ h_new,
                      float* __restrict__ x_pool,
                      int* __restrict__ batch_pool) {
    __shared__ float sc[NODES_PER_GRAPH];   // 32 KB
    const int g   = blockIdx.x;
    const int tid = threadIdx.x;
    for (int i = tid; i < NODES_PER_GRAPH; i += blockDim.x)
        sc[i] = score[g * NODES_PER_GRAPH + i];
    __syncthreads();

    for (int i = tid; i < NODES_PER_GRAPH; i += blockDim.x) {
        const float si = sc[i];
        int rank = 0;
        for (int j = 0; j < NODES_PER_GRAPH; ++j) {
            float sj = sc[j];
            rank += (sj > si) || ((sj == si) && (j < i));
        }
        if (rank < K_OUT) {
            const int node = g * NODES_PER_GRAPH + i;
            const int row  = g * K_OUT + rank;
            const float4* hv = (const float4*)(h_new + (size_t)node * HID);
            float4* ov = (float4*)(x_pool + (size_t)row * HID);
            #pragma unroll
            for (int c = 0; c < HID / 4; ++c) {
                float4 v = hv[c];
                v.x *= si; v.y *= si; v.z *= si; v.w *= si;
                ov[c] = v;
            }
            batch_pool[row] = g;
        }
    }
}

// ---------------------------------------------------------------
// host launcher
// ---------------------------------------------------------------
extern "C" void kernel_launch(void* const* d_in, const int* in_sizes, int n_in,
                              void* d_out, int out_size, void* d_ws, size_t ws_size,
                              hipStream_t stream) {
    const float* x        = (const float*)d_in[0];
    const float* mesh_pos = (const float*)d_in[1];
    const int*   edges    = (const int*)  d_in[2];
    // d_in[3] = batch_idx (derivable, unused)
    const float* proj_w   = (const float*)d_in[4];
    const float* proj_b   = (const float*)d_in[5];
    const float* msg_w1   = (const float*)d_in[6];
    const float* msg_b1   = (const float*)d_in[7];
    const float* msg_w2   = (const float*)d_in[8];
    const float* msg_b2   = (const float*)d_in[9];
    const float* upd_w1   = (const float*)d_in[10];
    const float* upd_b1   = (const float*)d_in[11];
    const float* upd_w2   = (const float*)d_in[12];
    const float* upd_b2   = (const float*)d_in[13];
    const float* rel_w    = (const float*)d_in[14];
    const float* rel_b    = (const float*)d_in[15];
    const float* root_w   = (const float*)d_in[16];

    // workspace layout (bytes)
    char* ws = (char*)d_ws;
    const size_t NH = (size_t)N_NODES * HID;             // 4,194,304
    float*  h      = (float*) (ws);                      // 16 MB
    float*  agg    = (float*) (ws + NH * 4);             // 16 MB
    float*  h_new  = (float*) (ws + NH * 8);             // 16 MB
    float*  nagg   = (float*) (ws + NH * 12);            // 16 MB
    __bf16* hbf    = (__bf16*)(ws + NH * 16);            //  8 MB
    __bf16* upd_in = (__bf16*)(ws + NH * 18);            // 16 MB
    float*  cnt    = (float*) (ws + NH * 22);            // 128 KB
    float*  score  = (float*) (ws + NH * 22 + N_NODES * 4);

    float* x_pool     = (float*)d_out;                       // 4*512*128 f32
    int*   batch_pool = (int*)((float*)d_out + (size_t)N_GRAPHS * K_OUT * HID);

    // 1) zero accumulators
    {
        size_t n = NH;
        int blk = 256;
        fill_kernel<<<(int)((n + blk - 1) / blk), blk, 0, stream>>>(agg, n, 0.0f);
        fill_kernel<<<(int)((n + blk - 1) / blk), blk, 0, stream>>>(nagg, n, 0.0f);
        fill_kernel<<<(N_NODES + 255) / 256, 256, 0, stream>>>(cnt, (size_t)N_NODES, 0.0f);
    }
    // 2) node features + positional embedding
    node_init_kernel<<<(int)(NH / 256), 256, 0, stream>>>(x, mesh_pos, proj_w, proj_b, h, hbf);
    // 3) in-degree
    degree_kernel<<<N_EDGES / 256, 256, 0, stream>>>(edges, cnt);
    // 4) edge message MLP + scatter (heavy WMMA kernel)
    edge_mlp_kernel<<<512, WAVES_PER_BLOCK * 32, 0, stream>>>(
        hbf, edges, msg_w1, msg_b1, msg_w2, msg_b2, agg);
    // 5) build update-MLP input
    node_prep_kernel<<<(int)(NH / 256), 256, 0, stream>>>(hbf, agg, cnt, upd_in);
    // 6) node update MLP (WMMA)
    update_mlp_kernel<<<256, WAVES_PER_BLOCK * 32, 0, stream>>>(
        upd_in, h, upd_w1, upd_b1, upd_w2, upd_b2, h_new);
    // 7) neighbor aggregation of updated features
    nagg_kernel<<<(int)(((size_t)N_EDGES * 32) / 256), 256, 0, stream>>>(edges, h_new, nagg);
    // 8) pooling scores
    score_kernel<<<N_NODES / 256, 256, 0, stream>>>(nagg, cnt, h_new, rel_w, rel_b, root_w, score);
    // 9) exact stable top-k + gather/scale
    topk_pool_kernel<<<N_GRAPHS, 1024, 0, stream>>>(score, h_new, x_pool, batch_pool);
}
